// ConditionalLatentFlowModelV2_15934328668508
// MI455X (gfx1250) — matrix operand order repaired and checked by reading, not compiled, not run on hardware
//
#include <hip/hip_runtime.h>
#include <hip/hip_bf16.h>
#include <math.h>

typedef __attribute__((ext_vector_type(16))) __bf16 v16bf;
typedef __attribute__((ext_vector_type(8)))  float  v8f;

#define B_    1024
#define D_    128
#define C_    256
#define H_    512
#define M_    64
#define FH_   2048   /* 4H */
#define HEADS 4
#define DH_   128
#define STEPS 4

__device__ __forceinline__ float gelu_exact(float x) {
  return 0.5f * x * (1.0f + erff(x * 0.70710678118654752f));
}

union Frag { v16bf v; uint4 u[2]; };

__device__ __forceinline__ v8f wmma_bf16(const Frag& a, const Frag& b, v8f c) {
  return __builtin_amdgcn_wmma_f32_16x16x32_bf16(false, a.v, false, b.v,
                                                 (short)0, c, false, false);
}

// ---------------------------------------------------------------------------
// bf16 WMMA GEMM:  C(R x N) = A(R x K) * W(K x N) + bias, with W pre-transposed
// to WT (N x K, bf16).  Each wave computes a 32x32 tile: 2 A fragments x 2 B
// fragments -> 4 v_wmma per 32-wide k-step (halves fragment traffic per FLOP
// vs a 16x16 tile).  Block = 4 waves tiling 64x64 of C.
// Epilogue fuses bias (+ optional exact GELU) and can emit fp32 (e.g. into a
// column slot of the fused z buffer) and/or bf16 (A operand of next GEMM).
// ---------------------------------------------------------------------------
__global__ __launch_bounds__(128) void gemm_bf16_wmma(
    const __bf16* __restrict__ A, const __bf16* __restrict__ WT,
    const float* __restrict__ bias,
    float* __restrict__ outF, int ldoF,
    __bf16* __restrict__ outB, int ldoB,
    int K, int act) {
  const int lane = threadIdx.x & 31;
  const int wave = threadIdx.x >> 5;
  const int row0 = blockIdx.y * 64 + (wave >> 1) * 32;
  const int col0 = blockIdx.x * 64 + (wave & 1) * 32;
  const int g = lane >> 4;        // lane group (0: lanes 0-15, 1: lanes 16-31)
  const int r = lane & 15;
  const __bf16* arow0 = A  + (size_t)(row0 + r) * K;
  const __bf16* arow1 = A  + (size_t)(row0 + 16 + r) * K;
  const __bf16* bcol0 = WT + (size_t)(col0 + r) * K;
  const __bf16* bcol1 = WT + (size_t)(col0 + 16 + r) * K;
  v8f c00 = {}, c01 = {}, c10 = {}, c11 = {};
  for (int kb = 0; kb < K; kb += 32) {
    Frag fa0, fa1, fb0, fb1;
    // 16-bit A 16x32 layout: g=0 -> K 0..7 / 16..23 ; g=1 -> K 8..15 / 24..31
    fa0.u[0] = *(const uint4*)(arow0 + kb + 8 * g);
    fa0.u[1] = *(const uint4*)(arow0 + kb + 16 + 8 * g);
    fa1.u[0] = *(const uint4*)(arow1 + kb + 8 * g);
    fa1.u[1] = *(const uint4*)(arow1 + kb + 16 + 8 * g);
    // 16-bit B 32x16 layout (same k order as A): g=0 -> K 0..7 / 16..23 ; g=1 -> K 8..15 / 24..31
    fb0.u[0] = *(const uint4*)(bcol0 + kb + 8 * g);
    fb0.u[1] = *(const uint4*)(bcol0 + kb + 16 + 8 * g);
    fb1.u[0] = *(const uint4*)(bcol1 + kb + 8 * g);
    fb1.u[1] = *(const uint4*)(bcol1 + kb + 16 + 8 * g);
    if (kb + 32 < K) {
      // near-scope prefetch of the next k-block (global_prefetch_b8)
      __builtin_prefetch(arow0 + kb + 32, 0, 3);
      __builtin_prefetch(arow1 + kb + 32, 0, 3);
      __builtin_prefetch(bcol0 + kb + 32, 0, 3);
      __builtin_prefetch(bcol1 + kb + 32, 0, 3);
    }
    c00 = wmma_bf16(fa0, fb0, c00);
    c01 = wmma_bf16(fa0, fb1, c01);
    c10 = wmma_bf16(fa1, fb0, c10);
    c11 = wmma_bf16(fa1, fb1, c11);
  }
  const float bv0 = bias[col0 + r];
  const float bv1 = bias[col0 + 16 + r];
#pragma unroll
  for (int i = 0; i < 8; ++i) {
    const int ra = row0 + i + 8 * g;        // C layout: VGPR i -> M=i (+8 g=1)
    const int rb = ra + 16;
    float v00 = c00[i] + bv0, v01 = c01[i] + bv1;
    float v10 = c10[i] + bv0, v11 = c11[i] + bv1;
    if (act == 1) {
      v00 = gelu_exact(v00); v01 = gelu_exact(v01);
      v10 = gelu_exact(v10); v11 = gelu_exact(v11);
    }
    if (outF) {
      outF[(size_t)ra * ldoF + col0 + r]      = v00;
      outF[(size_t)ra * ldoF + col0 + 16 + r] = v01;
      outF[(size_t)rb * ldoF + col0 + r]      = v10;
      outF[(size_t)rb * ldoF + col0 + 16 + r] = v11;
    }
    if (outB) {
      outB[(size_t)ra * ldoB + col0 + r]      = (__bf16)v00;
      outB[(size_t)ra * ldoB + col0 + 16 + r] = (__bf16)v01;
      outB[(size_t)rb * ldoB + col0 + r]      = (__bf16)v10;
      outB[(size_t)rb * ldoB + col0 + 16 + r] = (__bf16)v11;
    }
  }
}

// LayerNorm rows of X (R x Dk) with per-feature gamma/beta, emit bf16.
__global__ __launch_bounds__(128) void ln_rows_bf16(
    const float* __restrict__ X, const float* __restrict__ gw,
    const float* __restrict__ bw, __bf16* __restrict__ out, int Dk) {
  const int row = blockIdx.x;
  const int tid = threadIdx.x;
  const float* x = X + (size_t)row * Dk;
  __shared__ float red[128];
  float s = 0.f;
  for (int i = tid; i < Dk; i += 128) s += x[i];
  red[tid] = s; __syncthreads();
  for (int o = 64; o > 0; o >>= 1) { if (tid < o) red[tid] += red[tid + o]; __syncthreads(); }
  const float mu = red[0] / (float)Dk;
  __syncthreads();
  float v = 0.f;
  for (int i = tid; i < Dk; i += 128) { float d = x[i] - mu; v += d * d; }
  red[tid] = v; __syncthreads();
  for (int o = 64; o > 0; o >>= 1) { if (tid < o) red[tid] += red[tid + o]; __syncthreads(); }
  const float rs = rsqrtf(red[0] / (float)Dk + 1e-5f);
  __bf16* orow = out + (size_t)row * Dk;
  for (int i = tid; i < Dk; i += 128)
    orow[i] = (__bf16)((x[i] - mu) * rs * gw[i] + bw[i]);
}

// Multi-head attention over precomputed bf16 K/V memory. One wave per (b,head).
__global__ __launch_bounds__(128) void attention_kernel(
    const float* __restrict__ q, const __bf16* __restrict__ kmem,
    const __bf16* __restrict__ vmem, __bf16* __restrict__ summ) {
  const int lane = threadIdx.x & 31;
  const int wave = threadIdx.x >> 5;
  const int unit = blockIdx.x * 4 + wave;       // 0 .. B*HEADS-1
  const int b = unit >> 2, hd = unit & 3;
  const float*  qrow = q    + (size_t)b * H_ + hd * DH_;
  const __bf16* kb0  = kmem + (size_t)b * M_ * H_ + hd * DH_;
  const __bf16* vb0  = vmem + (size_t)b * M_ * H_ + hd * DH_;
  const __bf16* kr0 = kb0 + (size_t)lane * H_;
  const __bf16* kr1 = kb0 + (size_t)(lane + 32) * H_;
  float l0 = 0.f, l1 = 0.f;
  for (int d = 0; d < DH_; d += 4) {
    const float q0 = qrow[d], q1 = qrow[d + 1], q2 = qrow[d + 2], q3 = qrow[d + 3];
    l0 += q0 * (float)kr0[d]     + q1 * (float)kr0[d + 1]
        + q2 * (float)kr0[d + 2] + q3 * (float)kr0[d + 3];
    l1 += q0 * (float)kr1[d]     + q1 * (float)kr1[d + 1]
        + q2 * (float)kr1[d + 2] + q3 * (float)kr1[d + 3];
  }
  const float scale = 0.08838834764831845f;     // 128^-0.5
  l0 *= scale; l1 *= scale;
  float mx = fmaxf(l0, l1);
  for (int o = 16; o > 0; o >>= 1) mx = fmaxf(mx, __shfl_xor(mx, o, 32));
  const float e0 = __expf(l0 - mx), e1 = __expf(l1 - mx);
  float sum = e0 + e1;
  for (int o = 16; o > 0; o >>= 1) sum += __shfl_xor(sum, o, 32);
  const float inv = 1.f / sum;
  __shared__ float attn_s[4][64];
  attn_s[wave][lane] = e0 * inv;
  attn_s[wave][lane + 32] = e1 * inv;
  __syncthreads();
  float a0 = 0.f, a1 = 0.f, a2 = 0.f, a3 = 0.f;
  const int dbase = lane * 4;
  for (int m = 0; m < M_; ++m) {
    const float a = attn_s[wave][m];
    const __bf16* vp = vb0 + (size_t)m * H_ + dbase;
    a0 += a * (float)vp[0]; a1 += a * (float)vp[1];
    a2 += a * (float)vp[2]; a3 += a * (float)vp[3];
  }
  __bf16* o = summ + (size_t)b * H_ + hd * DH_ + dbase;
  o[0] = (__bf16)a0; o[1] = (__bf16)a1; o[2] = (__bf16)a2; o[3] = (__bf16)a3;
}

// te_row[j] = sinusoidal_embed(t) @ Wt + bt   (t is a broadcast scalar)
__global__ void time_proj_row(const float* __restrict__ Wt,
                              const float* __restrict__ bt,
                              float* __restrict__ te_row, float t) {
  const int j = blockIdx.x * blockDim.x + threadIdx.x;
  if (j >= H_) return;
  float acc = bt[j];
  const float nl = -logf(10000.f) / 255.f;
  for (int i = 0; i < H_; ++i) {
    const int f = i & 255;
    const float ang = t * __expf(nl * (float)f);
    const float e = (i < 256) ? __sinf(ang) : __cosf(ang);
    acc += e * Wt[(size_t)i * H_ + j];
  }
  te_row[j] = acc;
}

__global__ void bcast_te(const float* __restrict__ te_row, float* __restrict__ z) {
  const int idx = blockIdx.x * blockDim.x + threadIdx.x;   // B*H threads
  const int b = idx >> 9, j = idx & 511;
  z[(size_t)b * FH_ + 3 * H_ + j] = te_row[j];
}

// Convert fp32 weight W (K x N) -> bf16 transposed WT (N x K).
__global__ void wt_convert(const float* __restrict__ W, __bf16* __restrict__ WT,
                           int K, int N) {
  const int idx = blockIdx.x * blockDim.x + threadIdx.x;
  if (idx >= K * N) return;
  const int k = idx / N, n = idx - k * N;
  WT[(size_t)n * K + k] = (__bf16)W[idx];
}

__global__ void cvt_bf16(const float* __restrict__ X, __bf16* __restrict__ Y, int n) {
  const int i = blockIdx.x * blockDim.x + threadIdx.x;
  if (i < n) Y[i] = (__bf16)X[i];
}

__global__ void copy_f(const float* __restrict__ a, float* __restrict__ b, int n) {
  const int i = blockIdx.x * blockDim.x + threadIdx.x;
  if (i < n) b[i] = a[i];
}

__global__ void axpy_f(const float* __restrict__ y, const float* __restrict__ k,
                       float alpha, float* __restrict__ o, int n) {
  const int i = blockIdx.x * blockDim.x + threadIdx.x;
  if (i < n) o[i] = y[i] + alpha * k[i];
}

__global__ void heun_combine(const float* __restrict__ y, const float* __restrict__ k1,
                             const float* __restrict__ k2, float hdt,
                             float* __restrict__ o, int n) {
  const int i = blockIdx.x * blockDim.x + threadIdx.x;
  if (i < n) o[i] = y[i] + hdt * (k1[i] + k2[i]);
}

extern "C" void kernel_launch(void* const* d_in, const int* in_sizes, int n_in,
                              void* d_out, int out_size, void* d_ws, size_t ws_size,
                              hipStream_t stream) {
  const float* noise   = (const float*)d_in[0];
  const float* context = (const float*)d_in[1];
  const float* support = (const float*)d_in[2];
  const float* mem_g = (const float*)d_in[4];
  const float* mem_b = (const float*)d_in[5];
  const float* Wa  = (const float*)d_in[6];  const float* ba  = (const float*)d_in[7];
  const float* Wq  = (const float*)d_in[8];  const float* bq  = (const float*)d_in[9];
  const float* Wk  = (const float*)d_in[10]; const float* bk  = (const float*)d_in[11];
  const float* Wv  = (const float*)d_in[12]; const float* bvv = (const float*)d_in[13];
  const float* Wo  = (const float*)d_in[14]; const float* bo  = (const float*)d_in[15];
  const float* pp_g = (const float*)d_in[16]; const float* pp_b = (const float*)d_in[17];
  const float* Wp1 = (const float*)d_in[18]; const float* bp1 = (const float*)d_in[19];
  const float* Wp2 = (const float*)d_in[20]; const float* bp2 = (const float*)d_in[21];
  const float* Wc  = (const float*)d_in[22]; const float* bc  = (const float*)d_in[23];
  const float* Wt  = (const float*)d_in[24]; const float* bt  = (const float*)d_in[25];
  const float* f_g = (const float*)d_in[26]; const float* f_b = (const float*)d_in[27];
  const float* Wf1 = (const float*)d_in[28]; const float* bf1 = (const float*)d_in[29];
  const float* Wf2 = (const float*)d_in[30]; const float* bf2 = (const float*)d_in[31];
  float* out = (float*)d_out;

  // --- bump allocator over d_ws (layout is static & deterministic) ---
  char* ws = (char*)d_ws;
  size_t off = 0;
  auto alloc = [&](size_t bytes) -> void* {
    off = (off + 255) & ~(size_t)255;
    void* p = ws + off;
    off += bytes;
    return p;
  };
  auto allocBF = [&](size_t e) { return (__bf16*)alloc(e * 2); };
  auto allocF  = [&](size_t e) { return (float*)alloc(e * 4); };

  // bf16 transposed weights
  __bf16* WaT  = allocBF((size_t)D_ * H_);
  __bf16* WqT  = allocBF((size_t)H_ * H_);
  __bf16* WkT  = allocBF((size_t)H_ * H_);
  __bf16* WvT  = allocBF((size_t)H_ * H_);
  __bf16* WoT  = allocBF((size_t)H_ * H_);
  __bf16* Wp1T = allocBF((size_t)D_ * H_);
  __bf16* Wp2T = allocBF((size_t)H_ * H_);
  __bf16* WcT  = allocBF((size_t)C_ * H_);
  __bf16* Wf1T = allocBF((size_t)FH_ * H_);
  __bf16* Wf2T = allocBF((size_t)H_ * D_);
  // activations
  __bf16* lnmem_bf = allocBF((size_t)B_ * M_ * D_);
  __bf16* enc_bf   = allocBF((size_t)B_ * M_ * H_);
  __bf16* k_bf     = allocBF((size_t)B_ * M_ * H_);
  __bf16* v_bf     = allocBF((size_t)B_ * M_ * H_);
  __bf16* ctx_bf   = allocBF((size_t)B_ * C_);
  float*  z        = allocF((size_t)B_ * FH_);
  __bf16* zn_bf    = allocBF((size_t)B_ * FH_);
  __bf16* yn_bf    = allocBF((size_t)B_ * D_);
  __bf16* h1_bf    = allocBF((size_t)B_ * H_);
  __bf16* h_bf     = allocBF((size_t)B_ * H_);
  float*  qf       = allocF((size_t)B_ * H_);
  __bf16* summ_bf  = allocBF((size_t)B_ * H_);
  __bf16* f1_bf    = allocBF((size_t)B_ * H_);
  float*  te_row   = allocF(H_);
  float*  y_cur    = allocF((size_t)B_ * D_);
  float*  y_tmp    = allocF((size_t)B_ * D_);
  float*  kv1      = allocF((size_t)B_ * D_);
  float*  kv2      = allocF((size_t)B_ * D_);

  auto convw = [&](const float* W, __bf16* WT, int K, int N) {
    int n = K * N;
    wt_convert<<<(n + 255) / 256, 256, 0, stream>>>(W, WT, K, N);
  };
  auto gemm = [&](const __bf16* A, const __bf16* WT, const float* bias,
                  float* oF, int ldoF, __bf16* oB, int ldoB,
                  int R, int K, int N, int act) {
    dim3 grid(N / 64, R / 64);                 // wave = 32x32 tile, block = 64x64
    gemm_bf16_wmma<<<grid, 128, 0, stream>>>(A, WT, bias, oF, ldoF, oB, ldoB, K, act);
  };

  // ---------------- precompute (solver-invariant) ----------------
  convw(Wa,  WaT,  D_, H_);   convw(Wq,  WqT,  H_, H_);
  convw(Wk,  WkT,  H_, H_);   convw(Wv,  WvT,  H_, H_);
  convw(Wo,  WoT,  H_, H_);   convw(Wp1, Wp1T, D_, H_);
  convw(Wp2, Wp2T, H_, H_);   convw(Wc,  WcT,  C_, H_);
  convw(Wf1, Wf1T, FH_, H_);  convw(Wf2, Wf2T, H_, D_);
  cvt_bf16<<<(B_ * C_ + 255) / 256, 256, 0, stream>>>(context, ctx_bf, B_ * C_);

  ln_rows_bf16<<<B_ * M_, 128, 0, stream>>>(support, mem_g, mem_b, lnmem_bf, D_);
  gemm(lnmem_bf, WaT, ba, nullptr, 0, enc_bf, H_, B_ * M_, D_, H_, 0);   // enc
  gemm(enc_bf, WkT, bk,  nullptr, 0, k_bf, H_, B_ * M_, H_, H_, 0);      // k_mem
  gemm(enc_bf, WvT, bvv, nullptr, 0, v_bf, H_, B_ * M_, H_, H_, 0);      // v_mem
  gemm(ctx_bf, WcT, bc, z + 2 * H_, FH_, nullptr, 0, B_, C_, H_, 0);     // ctx -> z slot2

  copy_f<<<(B_ * D_ + 255) / 256, 256, 0, stream>>>(noise, y_cur, B_ * D_);

  // ---------------- one velocity evaluation ----------------
  auto velocity = [&](const float* yin, float t, float* kout) {
    ln_rows_bf16<<<B_, 128, 0, stream>>>(yin, pp_g, pp_b, yn_bf, D_);
    gemm(yn_bf, Wp1T, bp1, nullptr, 0, h1_bf, H_, B_, D_, H_, 1);        // GELU
    gemm(h1_bf, Wp2T, bp2, z, FH_, h_bf, H_, B_, H_, H_, 0);             // h -> z slot0
    gemm(h_bf, WqT, bq, qf, H_, nullptr, 0, B_, H_, H_, 0);              // q
    attention_kernel<<<B_ * HEADS / 4, 128, 0, stream>>>(qf, k_bf, v_bf, summ_bf);
    gemm(summ_bf, WoT, bo, z + H_, FH_, nullptr, 0, B_, H_, H_, 0);      // m -> z slot1
    time_proj_row<<<2, 256, 0, stream>>>(Wt, bt, te_row, t);
    bcast_te<<<B_ * H_ / 256, 256, 0, stream>>>(te_row, z);              // te -> z slot3
    ln_rows_bf16<<<B_, 128, 0, stream>>>(z, f_g, f_b, zn_bf, FH_);
    gemm(zn_bf, Wf1T, bf1, nullptr, 0, f1_bf, H_, B_, FH_, H_, 1);       // GELU
    gemm(f1_bf, Wf2T, bf2, kout, D_, nullptr, 0, B_, H_, D_, 0);         // velocity
  };

  // ---------------- Heun integration (num_steps = 4) ----------------
  const float dt = 1.0f / (float)STEPS;
  const int n = B_ * D_;
  for (int s = 0; s < STEPS; ++s) {
    const float t0 = (float)s * dt;
    velocity(y_cur, t0, kv1);
    axpy_f<<<(n + 255) / 256, 256, 0, stream>>>(y_cur, kv1, dt, y_tmp, n);
    velocity(y_tmp, t0 + dt, kv2);
    float* dst = (s == STEPS - 1) ? out : y_cur;
    heun_combine<<<(n + 255) / 256, 256, 0, stream>>>(y_cur, kv1, kv2, 0.5f * dt, dst, n);
  }
}
